// FastAttention_55748675502417
// MI455X (gfx1250) — compile-verified, matrix operand order checked by reading, not run on hardware
//
#include <hip/hip_runtime.h>

// ---------------------------------------------------------------------------
// FAVOR+ conditioned causal linear attention for MI455X (gfx1250, wave32).
// All heavy matmuls use v_wmma_f32_16x16x32_f16 with f32 accumulation.
// Contiguous global->LDS staging uses GLOBAL_LOAD_ASYNC_TO_LDS_B128 when the
// toolchain exposes the builtin (ASYNCcnt-tracked DMA path).
// ---------------------------------------------------------------------------

typedef __attribute__((ext_vector_type(16))) _Float16 v16h;
typedef __attribute__((ext_vector_type(8)))  float    v8f;
typedef __attribute__((ext_vector_type(4)))  int      v4i;

#define DEVFN static __device__ __forceinline__

// Problem constants (fixed by the reference: b=2,h=8,n=4096,d=64,m=256)
constexpr int BH   = 16;       // b*h
constexpr int N    = 4096;     // sequence length
constexpr int D    = 64;       // head dim
constexpr int M    = 256;      // projection features
constexpr int E    = 64;       // value dim
constexpr int L    = 256;      // COND_LEN
constexpr int NC   = 128;      // N_CHUNKS
constexpr int LC   = 30;       // chunk length = (4096-256)/128
constexpr int ROWS = BH * N;   // 65536

constexpr float NORMALIZER = 0.35355339059327373f;  // 64^-0.25
constexpr float RATIO      = 0.0625f;               // 256^-0.5
constexpr float EPS_K      = 1e-4f;
constexpr float EPS_C      = 1e-6f;

// Workspace layout (bytes)
constexpr size_t OFF_QP16  = 0;                                 // 32 MB  f16 qp
constexpr size_t OFF_KP16  = OFF_QP16 + (size_t)ROWS * M * 2;   // 32 MB  f16 kp
constexpr size_t OFF_CTX   = OFF_KP16 + (size_t)ROWS * M * 2;   // 128 MB f32 chunk ctx (also dash_k temp)
constexpr size_t OFF_KSUM  = OFF_CTX  + (size_t)BH * NC * M * E * 4; // 2 MB
constexpr size_t OFF_BASES = OFF_KSUM + (size_t)BH * NC * M * 4;     // 1 MB cond ctx
constexpr size_t OFF_BASEZ = OFF_BASES + (size_t)BH * M * E * 4;     // 16 KB cond ksum
constexpr size_t OFF_DIAG  = OFF_BASEZ + (size_t)BH * M * 4;         // 256 KB k diag
constexpr size_t OFF_GMAX  = OFF_DIAG  + (size_t)ROWS * 4;           // 4 B

// ---- async global->LDS copy (16B per lane), with VGPR-path fallback -------
#if defined(__gfx1250__) && __has_builtin(__builtin_amdgcn_global_load_async_to_lds_b128)
#define HAVE_ASYNC_LDS 1
#else
#define HAVE_ASYNC_LDS 0
#endif

DEVFN void async_copy_b128(const void* gsrc, void* ldst) {
#if HAVE_ASYNC_LDS
  __builtin_amdgcn_global_load_async_to_lds_b128(
      (__attribute__((address_space(1))) v4i*)gsrc,
      (__attribute__((address_space(3))) v4i*)ldst, 0, 0);
#else
  *(v4i*)ldst = *(const v4i*)gsrc;
#endif
}
DEVFN void async_wait0() {
#if HAVE_ASYNC_LDS
#if __has_builtin(__builtin_amdgcn_s_wait_asynccnt)
  __builtin_amdgcn_s_wait_asynccnt(0);
#else
  asm volatile("s_wait_asynccnt 0" ::: "memory");
#endif
#endif
}

// ---- order-preserving float <-> uint for atomicMax on floats --------------
DEVFN unsigned f2key(float x) {
  unsigned u = __float_as_uint(x);
  return (u & 0x80000000u) ? ~u : (u | 0x80000000u);
}
DEVFN float key2f(unsigned k) {
  unsigned u = (k & 0x80000000u) ? (k & 0x7FFFFFFFu) : ~k;
  return __uint_as_float(u);
}

// ---- WMMA fragment helpers (CDNA5 16x16x32 f16, wave32 layouts) -----------
// A operand from LDS row-major [M][K] (stride elems), covers K=0..31 at base.
// Same loader serves the B operand when LDS holds B^T row-major [N][K].
DEVFN v16h frag_ld(const _Float16* base, int stride, int lane) {
  const int r  = lane & 15;
  const int kb = (lane & 16) ? 8 : 0;
  const _Float16* p = base + r * stride;
  v16h f;
#pragma unroll
  for (int vv = 0; vv < 8; ++vv) {
    const int k0 = (vv < 4) ? (kb + 2 * vv) : (16 + kb + 2 * (vv - 4));
    f[2 * vv]     = p[k0];
    f[2 * vv + 1] = p[k0 + 1];
  }
  return f;
}
DEVFN v8f wmma16(v16h a, v16h b, v8f c) {
  return __builtin_amdgcn_wmma_f32_16x16x32_f16(false, a, false, b,
                                                (short)0, c, false, false);
}

// ===========================================================================
// K0: init global-max accumulator to encode(-inf)
// ===========================================================================
__global__ void k_init(unsigned* gmax) { *gmax = 0x007FFFFFu; }

// ===========================================================================
// K1a: qp = ratio*(exp(dash - diag - rowmax) + eps), dash via WMMA.
// grid = ROWS/16 blocks of 128 threads (4 waves); each block = 16 rows.
// ===========================================================================
__global__ __launch_bounds__(128) void k_proj_q(const float* __restrict__ q,
                                                const float* __restrict__ proj,
                                                _Float16* __restrict__ qp16) {
  extern __shared__ char smem[];
  _Float16* projL = (_Float16*)smem;                          // [256][64]
  _Float16* aL    = (_Float16*)(smem + 32768);                // [16][64]
  float*    dashL = (float*)(smem + 32768 + 2048);            // [16][256]
  float*    diagL = (float*)(smem + 32768 + 2048 + 16384);    // [16]
  float*    redL  = (float*)(smem + 32768 + 2048 + 16384 + 64); // [128]

  const int t = threadIdx.x, lane = t & 31, w = t >> 5;
  const size_t r0 = (size_t)blockIdx.x * 16;

  if (t < 16) diagL[t] = 0.0f;
  for (int i = t; i < M * D; i += 128) projL[i] = (_Float16)proj[i];
  __syncthreads();
  {
    const int row = t >> 3, c0 = (t & 7) * 8;
    float ss = 0.f;
#pragma unroll
    for (int i = 0; i < 8; ++i) {
      float x = q[(r0 + row) * D + c0 + i];
      ss += x * x;
      aL[row * D + c0 + i] = (_Float16)(NORMALIZER * x);
    }
    atomicAdd(&diagL[row], 0.5f * NORMALIZER * NORMALIZER * ss);
  }
  __syncthreads();

  const v16h a0 = frag_ld(aL, D, lane);
  const v16h a1 = frag_ld(aL + 32, D, lane);
#pragma unroll
  for (int i = 0; i < 4; ++i) {
    const int nt = w + 4 * i;
    v16h b0 = frag_ld(projL + nt * 16 * D, D, lane);
    v16h b1 = frag_ld(projL + nt * 16 * D + 32, D, lane);
    v8f c = {};
    c = wmma16(a0, b0, c);
    c = wmma16(a1, b1, c);
    const int ncol = lane & 15, mb = (lane & 16) ? 8 : 0;
#pragma unroll
    for (int vv = 0; vv < 8; ++vv)
      dashL[(mb + vv) * M + nt * 16 + ncol] = c[vv];
  }
  __syncthreads();
  {
    const int row = t >> 3, c0 = (t & 7) * 32;
    float mx = -3.4e38f;
#pragma unroll
    for (int i = 0; i < 32; ++i) mx = fmaxf(mx, dashL[row * M + c0 + i]);
    redL[t] = mx;
  }
  __syncthreads();
  if ((t & 7) == 0) {
    float mx = redL[t];
#pragma unroll
    for (int j = 1; j < 8; ++j) mx = fmaxf(mx, redL[t + j]);
    redL[t] = mx;
  }
  __syncthreads();
  {
    const int row = t >> 3, c0 = (t & 7) * 32;
    const float mx = redL[t & ~7];
    const float dg = diagL[row];
#pragma unroll
    for (int i = 0; i < 32; ++i) {
      float val = RATIO * (__expf(dashL[row * M + c0 + i] - dg - mx) + EPS_K);
      qp16[(r0 + row) * M + c0 + i] = (_Float16)val;
    }
  }
}

// ===========================================================================
// K1b: dash_k via WMMA -> f32 workspace, per-row diag, global max (atomic).
// ===========================================================================
__global__ __launch_bounds__(128) void k_proj_k(const float* __restrict__ kk,
                                                const float* __restrict__ proj,
                                                float* __restrict__ dashG,
                                                float* __restrict__ diagG,
                                                unsigned* __restrict__ gmax) {
  extern __shared__ char smem[];
  _Float16* projL = (_Float16*)smem;                // [256][64]
  _Float16* aL    = (_Float16*)(smem + 32768);      // [16][64]
  float*    diagL = (float*)(smem + 32768 + 2048);  // [16]
  float*    redL  = (float*)(smem + 32768 + 2048 + 64); // [128]

  const int t = threadIdx.x, lane = t & 31, w = t >> 5;
  const size_t r0 = (size_t)blockIdx.x * 16;

  if (t < 16) diagL[t] = 0.0f;
  for (int i = t; i < M * D; i += 128) projL[i] = (_Float16)proj[i];
  __syncthreads();
  {
    const int row = t >> 3, c0 = (t & 7) * 8;
    float ss = 0.f;
#pragma unroll
    for (int i = 0; i < 8; ++i) {
      float x = kk[(r0 + row) * D + c0 + i];
      ss += x * x;
      aL[row * D + c0 + i] = (_Float16)(NORMALIZER * x);
    }
    atomicAdd(&diagL[row], 0.5f * NORMALIZER * NORMALIZER * ss);
  }
  __syncthreads();
  if (t < 16) diagG[r0 + t] = diagL[t];

  const v16h a0 = frag_ld(aL, D, lane);
  const v16h a1 = frag_ld(aL + 32, D, lane);
  float mx = -3.4e38f;
#pragma unroll
  for (int i = 0; i < 4; ++i) {
    const int nt = w + 4 * i;
    v16h b0 = frag_ld(projL + nt * 16 * D, D, lane);
    v16h b1 = frag_ld(projL + nt * 16 * D + 32, D, lane);
    v8f c = {};
    c = wmma16(a0, b0, c);
    c = wmma16(a1, b1, c);
    const int ncol = lane & 15, mb = (lane & 16) ? 8 : 0;
#pragma unroll
    for (int vv = 0; vv < 8; ++vv) {
      dashG[(r0 + mb + vv) * M + nt * 16 + ncol] = c[vv];
      mx = fmaxf(mx, c[vv]);
    }
  }
  redL[t] = mx;
  __syncthreads();
  if (t == 0) {
    float m2 = redL[0];
    for (int j = 1; j < 128; ++j) m2 = fmaxf(m2, redL[j]);
    atomicMax(gmax, f2key(m2));
  }
}

// ===========================================================================
// K1c: kp = ratio*(exp(dash - diag - gmax) + eps)   (elementwise, f16 out)
// ===========================================================================
__global__ __launch_bounds__(256) void k_exp_k(const float* __restrict__ dashG,
                                               const float* __restrict__ diagG,
                                               const unsigned* __restrict__ gmaxU,
                                               _Float16* __restrict__ kp16) {
  const size_t idx = (size_t)blockIdx.x * 256 + threadIdx.x;
  const float mx = key2f(*gmaxU);
  const size_t r = idx >> 8;
  kp16[idx] = (_Float16)(RATIO * (__expf(dashG[idx] - diagG[r] - mx) + EPS_K));
}

// ===========================================================================
// K2: condition block. 16 blocks (one per bh) x 256 threads (8 waves).
//   a) ksum = sum_{rows<L} kp           -> baseZ
//   b) ctx  = Kcond^T * Vcond (WMMA)    -> baseS (+f16 copy ctxT in LDS)
//   c) out[0..255] = (Q * ctx) * dinv   (WMMA)
// ===========================================================================
__global__ __launch_bounds__(256) void k_cond(const _Float16* __restrict__ qp16,
                                              const _Float16* __restrict__ kp16,
                                              const float* __restrict__ v,
                                              float* __restrict__ baseS,
                                              float* __restrict__ baseZ,
                                              float* __restrict__ out) {
  extern __shared__ char smem[];
  _Float16* ctxT  = (_Float16*)smem;                  // [64][256] f16  (B^T for phase c)
  float*    ksumL = (float*)(smem + 32768);           // [256]
  float*    dinvL = (float*)(smem + 32768 + 1024);    // [256]
  char*     X     = smem + 32768 + 2048;              // overlay region
  _Float16* kpT   = (_Float16*)X;                     // [256][32] f16 (phase b)
  _Float16* vT    = (_Float16*)(X + 16384);           // [64][32]  f16 (phase b)
  _Float16* qpL   = (_Float16*)X;                     // [256][256] f16 (phase c)

  const int t = threadIdx.x, lane = t & 31, w = t >> 5;
  const int bh = blockIdx.x;
  const size_t rowBase = (size_t)bh * N;

  // ---- a) ksum over 256 condition rows
  {
    float acc = 0.f;
    for (int rr = 0; rr < L; ++rr)
      acc += (float)kp16[(rowBase + rr) * M + t];
    ksumL[t] = acc;
    baseZ[(size_t)bh * M + t] = acc;
  }

  // ---- b) ctx = K^T V  (M=256, N=64, K=256 in 8 steps of 32)
  v8f acc[2][4];
#pragma unroll
  for (int a = 0; a < 2; ++a)
#pragma unroll
    for (int b = 0; b < 4; ++b) acc[a][b] = (v8f){};

  for (int kc = 0; kc < 8; ++kc) {
    __syncthreads();
    for (int i = t; i < 256 * 32; i += 256) {
      const int mi = i >> 5, rrl = i & 31;
      kpT[mi * 32 + rrl] = kp16[(rowBase + kc * 32 + rrl) * M + mi];
    }
    for (int i = t; i < 64 * 32; i += 256) {
      const int ei = i >> 5, rrl = i & 31;
      vT[ei * 32 + rrl] = (_Float16)v[(rowBase + kc * 32 + rrl) * E + ei];
    }
    __syncthreads();
    v16h bfr[4];
#pragma unroll
    for (int nt = 0; nt < 4; ++nt) bfr[nt] = frag_ld(vT + nt * 16 * 32, 32, lane);
#pragma unroll
    for (int mtl = 0; mtl < 2; ++mtl) {
      v16h a = frag_ld(kpT + (2 * w + mtl) * 16 * 32, 32, lane);
#pragma unroll
      for (int nt = 0; nt < 4; ++nt) acc[mtl][nt] = wmma16(a, bfr[nt], acc[mtl][nt]);
    }
  }
  __syncthreads();
  {
    const int ncol = lane & 15, mb = (lane & 16) ? 8 : 0;
#pragma unroll
    for (int mtl = 0; mtl < 2; ++mtl)
#pragma unroll
      for (int nt = 0; nt < 4; ++nt)
#pragma unroll
        for (int vv = 0; vv < 8; ++vv) {
          const int mi = (2 * w + mtl) * 16 + mb + vv;
          const int ei = nt * 16 + ncol;
          const float val = acc[mtl][nt][vv];
          baseS[(size_t)bh * M * E + mi * E + ei] = val;
          ctxT[ei * 256 + mi] = (_Float16)val;
        }
  }
  __syncthreads();

  // ---- c) stage qp rows (contiguous 128 KB: async DMA to LDS), dinv, out
  {
    const char* gsrc = (const char*)(qp16 + rowBase * M);
    char* ldst = (char*)qpL;
    for (int i = t; i < (L * M * 2) / 16; i += 256)
      async_copy_b128(gsrc + (size_t)i * 16, ldst + (size_t)i * 16);
    async_wait0();
  }
  __syncthreads();
  {
    float acc1 = 0.f;
    for (int m = 0; m < M; ++m) acc1 += (float)qpL[t * M + m] * ksumL[m];
    dinvL[t] = 1.0f / acc1;
  }
  __syncthreads();
#pragma unroll
  for (int rtl = 0; rtl < 2; ++rtl) {
    const int rt = 2 * w + rtl;
    v8f oc[4];
#pragma unroll
    for (int nt = 0; nt < 4; ++nt) oc[nt] = (v8f){};
#pragma unroll
    for (int kc = 0; kc < 8; ++kc) {
      v16h a = frag_ld(qpL + rt * 16 * M + kc * 32, M, lane);
#pragma unroll
      for (int nt = 0; nt < 4; ++nt) {
        v16h b = frag_ld(ctxT + nt * 16 * M + kc * 32, M, lane);
        oc[nt] = wmma16(a, b, oc[nt]);
      }
    }
    const int ncol = lane & 15, mb = (lane & 16) ? 8 : 0;
#pragma unroll
    for (int nt = 0; nt < 4; ++nt)
#pragma unroll
      for (int vv = 0; vv < 8; ++vv) {
        const int row = rt * 16 + mb + vv;
        out[(rowBase + row) * E + nt * 16 + ncol] = oc[nt][vv] * dinvL[row];
      }
  }
}

// ===========================================================================
// K3: per-chunk sums.  2048 blocks (bh*chunk) x 128 threads (4 waves).
//   ksum_c[m], ctx_c = Kc^T Vc  (K=30 padded to 32, all WMMA)
// ===========================================================================
__global__ __launch_bounds__(128) void k_chunksum(const _Float16* __restrict__ kp16,
                                                  const float* __restrict__ v,
                                                  float* __restrict__ ctxG,
                                                  float* __restrict__ ksumG) {
  extern __shared__ char smem[];
  _Float16* kpT = (_Float16*)smem;            // [256][32]
  _Float16* vT  = (_Float16*)(smem + 16384);  // [64][32]

  const int t = threadIdx.x, lane = t & 31, w = t >> 5;
  const int bh = blockIdx.x / NC, c = blockIdx.x % NC;
  const size_t row0 = (size_t)bh * N + L + (size_t)c * LC;
  const size_t chunk = (size_t)bh * NC + c;

  for (int i = t; i < 256 * 32; i += 128) {
    const int mi = i >> 5, rrl = i & 31;
    kpT[mi * 32 + rrl] = (rrl < LC) ? kp16[(row0 + rrl) * M + mi] : (_Float16)0.0f;
  }
  for (int i = t; i < 64 * 32; i += 128) {
    const int ei = i >> 5, rrl = i & 31;
    vT[ei * 32 + rrl] = (rrl < LC) ? (_Float16)v[(row0 + rrl) * E + ei] : (_Float16)0.0f;
  }
  __syncthreads();

  // ksum (2 features per thread)
#pragma unroll
  for (int j = 0; j < 2; ++j) {
    const int mi = t + j * 128;
    float acc = 0.f;
#pragma unroll
    for (int rr = 0; rr < LC; ++rr) acc += (float)kpT[mi * 32 + rr];
    ksumG[chunk * M + mi] = acc;
  }

  v16h bfr[4];
#pragma unroll
  for (int nt = 0; nt < 4; ++nt) bfr[nt] = frag_ld(vT + nt * 16 * 32, 32, lane);
  const int ncol = lane & 15, mb = (lane & 16) ? 8 : 0;
#pragma unroll
  for (int mtl = 0; mtl < 4; ++mtl) {
    const int mt = 4 * w + mtl;
    v16h a = frag_ld(kpT + mt * 16 * 32, 32, lane);
#pragma unroll
    for (int nt = 0; nt < 4; ++nt) {
      v8f cacc = {};
      cacc = wmma16(a, bfr[nt], cacc);
#pragma unroll
      for (int vv = 0; vv < 8; ++vv) {
        const int mi = mt * 16 + mb + vv;
        const int ei = nt * 16 + ncol;
        ctxG[chunk * (M * E) + mi * E + ei] = cacc[vv];
      }
    }
  }
}

// ===========================================================================
// K4: in-place exclusive prefix scan over chunks, element-parallel.
//   data[bh][c][Eelem] <- base[bh][Eelem] + sum_{c'<c} data[bh][c'][Eelem]
// ===========================================================================
__global__ __launch_bounds__(256) void k_scan(float* __restrict__ data,
                                              const float* __restrict__ base,
                                              int Eelem) {
  const int idx = blockIdx.x * 256 + threadIdx.x;
  const int bh = idx / Eelem, e2 = idx % Eelem;
  float accv = base[(size_t)bh * Eelem + e2];
  float* p = data + (size_t)bh * NC * Eelem + e2;
  for (int c = 0; c < NC; ++c) {
    const float old = p[(size_t)c * Eelem];
    p[(size_t)c * Eelem] = accv;
    accv += old;
  }
}

// ===========================================================================
// K5: per-chunk output. 2048 blocks x 128 threads (4 waves).
//   numer = Q*S_carry + tril(Q*K^T)*V ;  den = q.Z + rowsum(tril) + eps*sum(q)
// ===========================================================================
__global__ __launch_bounds__(128) void k_chunkout(const _Float16* __restrict__ qp16,
                                                  const _Float16* __restrict__ kp16,
                                                  const float* __restrict__ v,
                                                  const float* __restrict__ ctxG,
                                                  const float* __restrict__ ksumG,
                                                  float* __restrict__ out) {
  extern __shared__ char smem[];
  _Float16* ST    = (_Float16*)smem;                        // [64][256]  S^T f16
  _Float16* qpL   = (_Float16*)(smem + 32768);              // [32][256]
  _Float16* kpL   = (_Float16*)(smem + 32768 + 16384);      // [32][256]
  _Float16* vT    = (_Float16*)(smem + 32768 + 32768);      // [64][32]
  _Float16* aLds  = (_Float16*)(smem + 32768 + 32768 + 4096);        // [32][32]
  float*    Z     = (float*)(smem + 32768 + 32768 + 4096 + 2048);    // [256]
  float*    rsumA = (float*)(smem + 32768 + 32768 + 4096 + 2048 + 1024); // [32]
  float*    dinvL = (float*)(smem + 32768 + 32768 + 4096 + 2048 + 1024 + 128); // [32]

  const int t = threadIdx.x, lane = t & 31, w = t >> 5;
  const int bh = blockIdx.x / NC, c = blockIdx.x % NC;
  const size_t row0 = (size_t)bh * N + L + (size_t)c * LC;
  const size_t chunk = (size_t)bh * NC + c;

  // ---- stage chunk rows of qp/kp via async LDS DMA (30 rows = 960 x 16B),
  //      pad rows 30..31 with zeros through the normal DS path.
  {
    const char* gq = (const char*)(qp16 + row0 * M);
    const char* gk = (const char*)(kp16 + row0 * M);
    for (int i = t; i < (LC * M * 2) / 16; i += 128) {
      async_copy_b128(gq + (size_t)i * 16, (char*)qpL + (size_t)i * 16);
      async_copy_b128(gk + (size_t)i * 16, (char*)kpL + (size_t)i * 16);
    }
    for (int i = LC * M + t; i < 32 * M; i += 128) {
      qpL[i] = (_Float16)0.0f;
      kpL[i] = (_Float16)0.0f;
    }
  }
  // ---- stage carries + transposed tiles (per-lane path)
#pragma unroll
  for (int j = 0; j < 2; ++j) Z[t + j * 128] = ksumG[chunk * M + t + j * 128];
  for (int i = t; i < M * E; i += 128) {
    const int mi = i >> 6, ei = i & 63;
    ST[ei * M + mi] = (_Float16)ctxG[chunk * (M * E) + i];
  }
  for (int i = t; i < 64 * 32; i += 128) {
    const int ei = i >> 5, rrl = i & 31;
    vT[ei * 32 + rrl] = (rrl < LC) ? (_Float16)v[(row0 + rrl) * E + ei] : (_Float16)0.0f;
  }
  if (t < 32) rsumA[t] = 0.0f;
  async_wait0();
  __syncthreads();

  const int ncol = lane & 15, mb = (lane & 16) ? 8 : 0;
  const int mt = w >> 1;           // output M-tile (rows)
  const int ntb = (w & 1) * 2;     // output N-tiles ntb, ntb+1

  // ---- 1) numer += Q * S_carry  (K=256)
  v8f oc[2] = {(v8f){}, (v8f){}};
#pragma unroll
  for (int kc = 0; kc < 8; ++kc) {
    v16h a = frag_ld(qpL + mt * 16 * M + kc * 32, M, lane);
#pragma unroll
    for (int j = 0; j < 2; ++j) {
      v16h b = frag_ld(ST + (ntb + j) * 16 * M + kc * 32, M, lane);
      oc[j] = wmma16(a, b, oc[j]);
    }
  }

  // ---- 2) A = Q K^T (this wave owns one 16x16 tile), mask tril, rowsums
  {
    const int mt2 = w >> 1, nt2 = w & 1;
    v8f ca = {};
#pragma unroll
    for (int kc = 0; kc < 8; ++kc) {
      v16h a = frag_ld(qpL + mt2 * 16 * M + kc * 32, M, lane);
      v16h b = frag_ld(kpL + nt2 * 16 * M + kc * 32, M, lane);
      ca = wmma16(a, b, ca);
    }
#pragma unroll
    for (int vv = 0; vv < 8; ++vv) {
      const int row = mt2 * 16 + mb + vv;
      const int col = nt2 * 16 + ncol;
      float val = (col <= row) ? ca[vv] : 0.0f;
      aLds[row * 32 + col] = (_Float16)val;
      // half-wave (16-lane) reduction over columns of this tile
#pragma unroll
      for (int sh = 1; sh < 16; sh <<= 1) val += __shfl_xor(val, sh, 32);
      if (ncol == 0) atomicAdd(&rsumA[row], val);
    }
  }
  __syncthreads();

  // ---- 4) denominator per row (threads 0..31)
  if (t < 32) {
    float a1 = 0.f, a2 = 0.f;
    for (int m = 0; m < M; ++m) {
      const float qv = (float)qpL[t * M + m];
      a1 += qv * Z[m];
      a2 += qv;
    }
    dinvL[t] = 1.0f / (a1 + rsumA[t] + EPS_C * a2);
  }

  // ---- 3) numer += tril(A) * V  (K=32, one step)
  {
    v16h a = frag_ld(aLds + mt * 16 * 32, 32, lane);
#pragma unroll
    for (int j = 0; j < 2; ++j) {
      v16h b = frag_ld(vT + (ntb + j) * 16 * 32, 32, lane);
      oc[j] = wmma16(a, b, oc[j]);
    }
  }
  __syncthreads();

  // ---- 5) write output rows < 30
#pragma unroll
  for (int j = 0; j < 2; ++j)
#pragma unroll
    for (int vv = 0; vv < 8; ++vv) {
      const int row = mt * 16 + mb + vv;
      if (row < LC)
        out[(row0 + row) * E + (ntb + j) * 16 + ncol] = oc[j][vv] * dinvL[row];
    }
}

// ===========================================================================
extern "C" void kernel_launch(void* const* d_in, const int* in_sizes, int n_in,
                              void* d_out, int out_size, void* d_ws, size_t ws_size,
                              hipStream_t stream) {
  (void)in_sizes; (void)n_in; (void)out_size; (void)ws_size;
  const float* q    = (const float*)d_in[0];
  const float* k    = (const float*)d_in[1];
  const float* v    = (const float*)d_in[2];
  const float* proj = (const float*)d_in[3];
  float* out = (float*)d_out;
  char*  ws  = (char*)d_ws;

  _Float16* qp16 = (_Float16*)(ws + OFF_QP16);
  _Float16* kp16 = (_Float16*)(ws + OFF_KP16);
  float*    ctxG = (float*)(ws + OFF_CTX);     // also dash_k temp
  float*    ksumG= (float*)(ws + OFF_KSUM);
  float*    baseS= (float*)(ws + OFF_BASES);
  float*    baseZ= (float*)(ws + OFF_BASEZ);
  float*    diagG= (float*)(ws + OFF_DIAG);
  unsigned* gmax = (unsigned*)(ws + OFF_GMAX);

  constexpr size_t SM_PROJQ = 32768 + 2048 + 16384 + 64 + 512;
  constexpr size_t SM_PROJK = 32768 + 2048 + 64 + 512;
  constexpr size_t SM_COND  = 32768 + 2048 + 131072;
  constexpr size_t SM_CSUM  = 16384 + 4096;
  constexpr size_t SM_COUT  = 32768 + 32768 + 4096 + 2048 + 1024 + 128 + 128;

  k_init<<<1, 1, 0, stream>>>(gmax);
  k_proj_q<<<ROWS / 16, 128, SM_PROJQ, stream>>>(q, proj, qp16);
  k_proj_k<<<ROWS / 16, 128, SM_PROJK, stream>>>(k, proj, ctxG /*dash temp*/, diagG, gmax);
  k_exp_k<<<(ROWS * (size_t)M) / 256, 256, 0, stream>>>(ctxG, diagG, gmax, kp16);
  k_cond<<<BH, 256, SM_COND, stream>>>(qp16, kp16, v, baseS, baseZ, out);
  k_chunksum<<<BH * NC, 128, SM_CSUM, stream>>>(kp16, v, ctxG, ksumG);
  k_scan<<<(BH * M * E) / 256, 256, 0, stream>>>(ctxG, baseS, M * E);
  k_scan<<<(BH * M) / 256, 256, 0, stream>>>(ksumG, baseZ, M);
  k_chunkout<<<BH * NC, 128, SM_COUT, stream>>>(qp16, kp16, v, ctxG, ksumG, out);
}
